// RPN_33646773797077
// MI455X (gfx1250) — matrix operand order repaired and checked by reading, not compiled, not run on hardware
//
#include <hip/hip_runtime.h>
#include <hip/hip_bf16.h>
#include <stdint.h>

// ---------------------------------------------------------------------------
// Shapes
// ---------------------------------------------------------------------------
#define B_      8
#define NPTS    1024
#define NGT     10
#define GTC     9          // gt_labels channels: dims(3)+ctrs(3)+th(1)+extra(2)
#define NANCH   5120       // 5 * NPTS
#define NSORT   8192       // pad 5120 -> pow2 for bitonic
#define MSEL    128        // NMIN + NMAX
#define NPOOL   64
#define EXTV    0.2f
#define K_RP1   288        // 259 padded to multiple of 32
#define ROWS_PN 8192       // B*NPTS
#define ROWS_RP 65536      // B*M*NPOOL

typedef __attribute__((ext_vector_type(16))) _Float16 v16h;
typedef __attribute__((ext_vector_type(8)))  _Float16 v8h;
typedef __attribute__((ext_vector_type(8)))  float    v8f;

union Frag16 { v16h v; v8h h[2]; };

// gfx1250 async global->LDS path (ASYNCcnt); guarded so absence of the
// builtin falls back to the direct-global WMMA pipeline.
#if __has_builtin(__builtin_amdgcn_global_load_async_to_lds_b128) && \
    __has_builtin(__builtin_amdgcn_s_wait_asynccnt)
#define HAVE_ASYNC_LDS 1
#else
#define HAVE_ASYNC_LDS 0
#endif

#if HAVE_ASYNC_LDS
typedef int vi4_ __attribute__((vector_size(16)));
__device__ __forceinline__ void issue_async_b128(const _Float16* gsrc,
                                                 _Float16* lds_dst) {
  // builtin signature: (v4i AS1*, v4i AS3*, imm offset, imm cpol)
  __builtin_amdgcn_global_load_async_to_lds_b128(
      (__attribute__((address_space(1))) vi4_*)gsrc,
      (__attribute__((address_space(3))) vi4_*)lds_dst,
      0, 0);
}
#endif

// NEW_CENTERS = _C @ _ROT -> {(0,0,1),(0,0,-1),(2.5,0,-1),(2.5,0,1),(2.5,0,0)}
__constant__ float c_cx[5] = {0.f, 0.f, 2.5f, 2.5f, 2.5f};
__constant__ float c_cz[5] = {1.f, -1.f, -1.f, 1.f, 0.f};

__device__ __forceinline__ unsigned mono_enc(float f) {
  unsigned u = __float_as_uint(f);
  return (u >> 31) ? ~u : (u | 0x80000000u);
}
__device__ __forceinline__ float mono_dec(unsigned k) {
  return __uint_as_float((k & 0x80000000u) ? (k & 0x7fffffffu) : ~k);
}

// ---------------------------------------------------------------------------
// Input packing: points (B,4,NPTS) f32 -> A0h (8192 x 32) f16 zero-padded,
// plus pts (B*NPTS,3) f32 side buffer (channels 0..2).
// ---------------------------------------------------------------------------
__global__ void k_pack_points(const float* __restrict__ pin,
                              _Float16* __restrict__ A0,
                              float* __restrict__ pts) {
  int r = blockIdx.x * blockDim.x + threadIdx.x;
  if (r >= ROWS_PN) return;
  int b = r / NPTS, i = r % NPTS;
#pragma unroll
  for (int c = 0; c < 32; ++c) {
    float v = (c < 4) ? pin[(size_t)b * 4 * NPTS + (size_t)c * NPTS + i] : 0.f;
    A0[(size_t)r * 32 + c] = (_Float16)v;
    if (c < 3) pts[(size_t)r * 3 + c] = v;
  }
}

// Weights (N x Kin) f32 -> (N x Kp) f16, zero pad K.
__global__ void k_convert_w(const float* __restrict__ Ws,
                            _Float16* __restrict__ Wd,
                            int N, int Kin, int Kp) {
  int idx = blockIdx.x * blockDim.x + threadIdx.x;
  if (idx >= N * Kp) return;
  int n = idx / Kp, k = idx % Kp;
  Wd[idx] = (_Float16)((k < Kin) ? Ws[(size_t)n * Kin + k] : 0.f);
}

// ---------------------------------------------------------------------------
// Shared WMMA accumulation for one 16x16 tile.
// A: (Mrows x K) f16 row-major, W: (N x K) f16 row-major, K % 32 == 0.
// Block = (32,4): all 4 waves share the same 16-row M strip (same mt) and
// stage the 16x32 A tile in LDS via double-buffered async global->LDS loads;
// each wave owns a distinct N tile and streams W directly (contiguous b128).
// Fragment layouts per CDNA5 ISA 7.12.2.
// ---------------------------------------------------------------------------
__device__ __forceinline__ v8f gemm_acc(const _Float16* __restrict__ A,
                                        const _Float16* __restrict__ W,
                                        int K, int mt, int nt,
                                        _Float16 (*atile)[16 * 32]) {
  const int lane = threadIdx.x;
  const int colB = nt * 16 + (lane & 15);
  const int kbA = (lane < 16) ? 0 : 8;    // A: hi-lanes carry K+8 block
  const int koB = (lane < 16) ? 0 : 8;    // B: hi-lanes carry K+8 block
  const _Float16* wp = W + (size_t)colB * K + koB;
  v8f acc = {};
#if HAVE_ASYNC_LDS
  const int tid = threadIdx.y * 32 + lane;   // 0..127: one 16B chunk each
  const int arow = tid >> 2, apart = tid & 3;
  const _Float16* ag = A + (size_t)(mt * 16 + arow) * K + apart * 8;
  const int aoff = tid * 8;                  // LDS halves offset of the chunk
  const int rowo = (lane & 15) * 32 + kbA;   // fragment base within tile
  const int T = K / 32;
  issue_async_b128(ag, &atile[0][aoff]);
  for (int i = 0; i < T; ++i) {
    const int kk = i * 32;
    if (i + 1 < T) {
      issue_async_b128(ag + kk + 32, &atile[(i + 1) & 1][aoff]);
      __builtin_amdgcn_s_wait_asynccnt(1);   // in-order: tile i has landed
    } else {
      __builtin_amdgcn_s_wait_asynccnt(0);
    }
    __syncthreads();                         // tile i visible to all 4 waves
    __builtin_prefetch(wp + kk + 256, 0, 3); // near-scope prefetch of W
    Frag16 fa, fb;
    fa.h[0] = *(const v8h*)(&atile[i & 1][rowo]);       // ds_load_b128
    fa.h[1] = *(const v8h*)(&atile[i & 1][rowo + 16]);
    fb.h[0] = *(const v8h*)(wp + kk);
    fb.h[1] = *(const v8h*)(wp + kk + 16);
    acc = __builtin_amdgcn_wmma_f32_16x16x32_f16(false, fa.v, false, fb.v,
                                                 (short)0, acc, false, false);
    __syncthreads();                         // reads done before overwrite
  }
#else
  (void)atile;
  const int rowA = mt * 16 + (lane & 15);
  const _Float16* ap = A + (size_t)rowA * K + kbA;
  for (int kk = 0; kk < K; kk += 32) {
    __builtin_prefetch(ap + kk + 128, 0, 3);
    __builtin_prefetch(wp + kk + 128, 0, 3);
    Frag16 fa, fb;
    fa.h[0] = *(const v8h*)(ap + kk);
    fa.h[1] = *(const v8h*)(ap + kk + 16);
    fb.h[0] = *(const v8h*)(wp + kk);
    fb.h[1] = *(const v8h*)(wp + kk + 16);
    acc = __builtin_amdgcn_wmma_f32_16x16x32_f16(false, fa.v, false, fb.v,
                                                 (short)0, acc, false, false);
  }
#endif
  return acc;
}

// out[m,n] = sum_k A[m,k]*W[n,k] + bias[n], stored f16.
__global__ __launch_bounds__(128)
void k_gemm_f16(const _Float16* __restrict__ A, const _Float16* __restrict__ W,
                const float* __restrict__ bias, _Float16* __restrict__ out,
                int K, int N) {
  __shared__ _Float16 atile[2][16 * 32];
  const int lane = threadIdx.x;
  const int mt = blockIdx.x;
  const int nt = blockIdx.y * 4 + threadIdx.y;
  v8f acc = gemm_acc(A, W, K, mt, nt, atile);
  const int colB = nt * 16 + (lane & 15);
  const float bv = bias ? bias[colB] : 0.f;
  const int rbase = mt * 16 + ((lane < 16) ? 0 : 8);
#pragma unroll
  for (int r = 0; r < 8; ++r)
    out[(size_t)(rbase + r) * N + colB] = (_Float16)(acc[r] + bv);
}

// rp2 GEMM: fused epilogue — per-channel sum/sumsq atomics (for global BN)
// and monotone-uint atomicMax over the 64-row pool axis (each 16-row tile is
// fully inside one box since 64 % 16 == 0). Avoids a 256MB f32 spill.
__global__ __launch_bounds__(128)
void k_gemm_rp2(const _Float16* __restrict__ A, const _Float16* __restrict__ W,
                const float* __restrict__ bias,
                float* __restrict__ chsum, float* __restrict__ chsq,
                unsigned* __restrict__ premax, int K, int N) {
  __shared__ _Float16 atile[2][16 * 32];
  const int lane = threadIdx.x;
  const int mt = blockIdx.x;
  const int nt = blockIdx.y * 4 + threadIdx.y;
  v8f acc = gemm_acc(A, W, K, mt, nt, atile);
  const int colB = nt * 16 + (lane & 15);
  const float bv = bias[colB];
  float s = 0.f, q = 0.f, mx = -3.4e38f;
#pragma unroll
  for (int r = 0; r < 8; ++r) {
    float v = acc[r] + bv;
    s += v; q += v * v; mx = fmaxf(mx, v);
  }
  // lanes l and l+16 hold the same output column; combine then 16 atomics.
  s += __shfl_down(s, 16, 32);
  q += __shfl_down(q, 16, 32);
  mx = fmaxf(mx, __shfl_down(mx, 16, 32));
  if (lane < 16) {
    atomicAdd(&chsum[colB], s);
    atomicAdd(&chsq[colB], q);
    int bm = (mt * 16) >> 6;              // box index for this tile
    atomicMax(&premax[(size_t)bm * N + colB], mono_enc(mx));
  }
}

// ---------------------------------------------------------------------------
// Per-channel stats over rows of an f16 (Mrows x N) matrix.
// ---------------------------------------------------------------------------
__global__ __launch_bounds__(256)
void k_colstats(const _Float16* __restrict__ X, float* __restrict__ sum,
                float* __restrict__ sq, int Mrows, int N) {
  int col = blockIdx.x;
  __shared__ float ss[256], qq[256];
  float s = 0.f, q = 0.f;
  for (int r = threadIdx.x; r < Mrows; r += 256) {
    float v = (float)X[(size_t)r * N + col];
    s += v; q += v * v;
  }
  ss[threadIdx.x] = s; qq[threadIdx.x] = q;
  __syncthreads();
  for (int st = 128; st > 0; st >>= 1) {
    if (threadIdx.x < st) {
      ss[threadIdx.x] += ss[threadIdx.x + st];
      qq[threadIdx.x] += qq[threadIdx.x + st];
    }
    __syncthreads();
  }
  if (threadIdx.x == 0) { sum[col] = ss[0]; sq[col] = qq[0]; }
}

__global__ void k_bn_apply(_Float16* __restrict__ X, const float* __restrict__ sum,
                           const float* __restrict__ sq, const float* __restrict__ g,
                           const float* __restrict__ be, int Mrows, int N, int relu) {
  size_t idx = (size_t)blockIdx.x * blockDim.x + threadIdx.x;
  size_t total = (size_t)Mrows * N;
  if (idx >= total) return;
  int col = (int)(idx % N);
  float inv = 1.f / (float)Mrows;
  float m = sum[col] * inv;
  float var = sq[col] * inv - m * m;
  float rs = rsqrtf(var + 1e-5f);
  float v = ((float)X[idx] - m) * rs * g[col] + be[col];
  if (relu) v = fmaxf(v, 0.f);
  X[idx] = (_Float16)v;
}

// ---------------------------------------------------------------------------
// BEV IoU (Sutherland-Hodgman clip of GT quad against anchor quad).
// b5 = [cx, cz, w, l, theta]
// ---------------------------------------------------------------------------
__device__ void bev_corners(const float* b, float* cx, float* cz) {
  const float lxs[4] = {0.5f, -0.5f, -0.5f, 0.5f};
  const float lzs[4] = {0.5f, 0.5f, -0.5f, -0.5f};
  float c = cosf(b[4]), s = sinf(b[4]);
#pragma unroll
  for (int k = 0; k < 4; ++k) {
    float lx = lxs[k] * b[2], lz = lzs[k] * b[3];
    cx[k] = b[0] + lx * c - lz * s;
    cz[k] = b[1] + lx * s + lz * c;
  }
}

__device__ float bev_iou(const float* g, const float* a) {
  float px[8], pz[8];
  float gx[4], gz[4], ax[4], az[4];
  bev_corners(g, gx, gz);
  bev_corners(a, ax, az);
  int cnt = 4;
#pragma unroll
  for (int k = 0; k < 4; ++k) { px[k] = gx[k]; pz[k] = gz[k]; }
  for (int e = 0; e < 4; ++e) {
    float e1x = ax[e], e1z = az[e];
    float e2x = ax[(e + 1) & 3], e2z = az[(e + 1) & 3];
    float ex = e2x - e1x, ez = e2z - e1z;
    float qx[8], qz[8];
    int m = 0;
    for (int i = 0; i < cnt; ++i) {
      int j = (i + 1) % cnt;
      float dc = ex * (pz[i] - e1z) - ez * (px[i] - e1x);
      float dn = ex * (pz[j] - e1z) - ez * (px[j] - e1x);
      bool ic = dc >= 0.f, in = dn >= 0.f;
      if (ic) { qx[m] = px[i]; qz[m] = pz[i]; ++m; }
      if (ic != in) {
        float den = dc - dn;
        if (fabsf(den) < 1e-9f) den = 1e-9f;
        float t = dc / den;
        qx[m] = px[i] + t * (px[j] - px[i]);
        qz[m] = pz[i] + t * (pz[j] - pz[i]);
        ++m;
      }
    }
    cnt = m;
    for (int i = 0; i < m; ++i) { px[i] = qx[i]; pz[i] = qz[i]; }
    if (cnt == 0) break;
  }
  float inter = 0.f;
  for (int i = 0; i < cnt; ++i) {
    int j = (i + 1) % cnt;
    inter += px[i] * pz[j] - px[j] * pz[i];
  }
  inter = 0.5f * fabsf(inter);
  float u = g[2] * g[3] + a[2] * a[3] - inter;
  return inter / fmaxf(u, 1e-8f);
}

__global__ void k_iou(const float* __restrict__ pts, const float* __restrict__ gt,
                      float* __restrict__ max_iou, int* __restrict__ max_ind) {
  int t = blockIdx.x * blockDim.x + threadIdx.x;
  if (t >= B_ * NANCH) return;
  int b = t / NANCH, a = t % NANCH;
  int p = a / 5, j = a % 5;
  const float* pp = pts + (size_t)(b * NPTS + p) * 3;
  float ab[5] = {pp[0] - c_cx[j], pp[2] - c_cz[j], 2.f, 5.f, 0.f};
  float best = -1.f;
  int bi = 0;
  for (int gi = 0; gi < NGT; ++gi) {
    const float* L = gt + (size_t)(b * NGT + gi) * GTC;
    // gbev = [ctr_x, ctr_z, dim0, dim2, theta]
    float gb[5] = {L[3], L[5], L[0], L[2], L[6]};
    float iou = bev_iou(gb, ab);
    if (iou > best) { best = iou; bi = gi; }
  }
  max_iou[t] = best;
  max_ind[t] = bi;
}

// ---------------------------------------------------------------------------
// Per-batch stable ascending argsort of 5120 IoUs via LDS bitonic sort on
// packed (monotone_float<<32 | index) keys; select bottom-64 + top-64.
// Writes sel_* outputs + a workspace copy of boxes for the pooling stage.
// ---------------------------------------------------------------------------
__global__ __launch_bounds__(512)
void k_sortsel(const float* __restrict__ max_iou, const int* __restrict__ max_ind,
               const float* __restrict__ pts,
               int* __restrict__ out_lab, float* __restrict__ out_iou,
               float* __restrict__ out_boxes, int* __restrict__ out_selidx,
               float* __restrict__ boxes_ws) {
  __shared__ unsigned long long keys[NSORT];   // 64 KB LDS
  const int b = blockIdx.x;
  const int tid = threadIdx.x;
  for (int t = tid; t < NSORT; t += 512) {
    unsigned hk = (t < NANCH) ? mono_enc(max_iou[(size_t)b * NANCH + t]) : 0xFFFFFFFFu;
    keys[t] = ((unsigned long long)hk << 32) | (unsigned)t;
  }
  __syncthreads();
  for (unsigned k = 2; k <= NSORT; k <<= 1) {
    for (unsigned j = k >> 1; j > 0; j >>= 1) {
      for (int t = tid; t < NSORT; t += 512) {
        unsigned ixj = (unsigned)t ^ j;
        if (ixj > (unsigned)t) {
          bool up = ((t & k) == 0);
          unsigned long long x = keys[t], y = keys[ixj];
          if ((x > y) == up) { keys[t] = y; keys[ixj] = x; }
        }
      }
      __syncthreads();
    }
  }
  for (int m = tid; m < MSEL; m += 512) {
    int pos = (m < 64) ? m : (NANCH - MSEL + m);   // order[:64] ++ order[-64:]
    int a = (int)(keys[pos] & 0xFFFFFFFFu);
    float iou = max_iou[(size_t)b * NANCH + a];
    int p = a / 5, j = a % 5;
    const float* pp = pts + (size_t)(b * NPTS + p) * 3;
    // a7 = [cent_x, cent_y, cent_z, DIM[1]=2, DIM[0]=2, DIM[2]=5, 0]
    float box[7] = {pp[0] - c_cx[j], pp[1], pp[2] - c_cz[j], 2.f, 2.f, 5.f, 0.f};
    int o = b * MSEL + m;
#pragma unroll
    for (int q = 0; q < 7; ++q) {
      out_boxes[(size_t)o * 7 + q] = box[q];
      boxes_ws[(size_t)o * 7 + q] = box[q];
    }
    out_iou[o] = iou;
    out_lab[o] = (iou > 0.5f) ? 1 : 0;
    out_selidx[o] = max_ind[(size_t)b * NANCH + a];
  }
}

// ---------------------------------------------------------------------------
// Pooling: per selected box, stable partition (inside first, original order),
// take 64, emit rows of [feat(256), pts-center(3), pad..288] as f16.
// ---------------------------------------------------------------------------
__global__ __launch_bounds__(256)
void k_pool(const float* __restrict__ boxes_ws, const float* __restrict__ pts,
            const _Float16* __restrict__ ftr, _Float16* __restrict__ pooled) {
  const int bm = blockIdx.x;          // 0..1023
  const int b = bm >> 7;
  __shared__ unsigned char inflag[NPTS];
  __shared__ short ord[NPOOL];
  __shared__ unsigned char vflag[NPOOL];
  __shared__ float bx[7];
  if (threadIdx.x < 7) bx[threadIdx.x] = boxes_ws[(size_t)bm * 7 + threadIdx.x];
  __syncthreads();
  float cth = cosf(bx[6]), sth = sinf(bx[6]);
  float hx = bx[4] * 0.5f + EXTV, hz = bx[5] * 0.5f + EXTV;
  for (int p = threadIdx.x; p < NPTS; p += 256) {
    const float* pp = pts + (size_t)(b * NPTS + p) * 3;
    float dx = pp[0] - bx[0], dz = pp[2] - bx[2];
    float xl = dx * cth + dz * sth;
    float zl = -dx * sth + dz * cth;
    bool ins = (fabsf(xl) < hx) && (fabsf(zl) < hz) &&
               (pp[1] > bx[1] - bx[3] - EXTV) && (pp[1] < bx[1] + EXTV);
    inflag[p] = ins ? 1 : 0;
  }
  __syncthreads();
  if (threadIdx.x == 0) {           // stable partition, first 64
    int m = 0;
    for (int p = 0; p < NPTS && m < NPOOL; ++p)
      if (inflag[p]) { ord[m] = (short)p; vflag[m] = 1; ++m; }
    for (int p = 0; p < NPTS && m < NPOOL; ++p)
      if (!inflag[p]) { ord[m] = (short)p; vflag[m] = 0; ++m; }
  }
  __syncthreads();
  for (int idx = threadIdx.x; idx < NPOOL * K_RP1; idx += 256) {
    int slot = idx / K_RP1, ch = idx % K_RP1;
    int p = ord[slot];
    bool valid = vflag[slot] != 0;
    float v;
    if (ch < 256) {
      v = valid ? (float)ftr[(size_t)(b * NPTS + p) * 256 + ch] : 0.f;
    } else if (ch < 259) {
      int c = ch - 256;
      float pv = pts[(size_t)(b * NPTS + p) * 3 + c];
      v = (valid ? pv : 0.f) - bx[c];   // center subtract applies to all slots
    } else {
      v = 0.f;                          // K padding
    }
    pooled[(size_t)(bm * NPOOL + slot) * K_RP1 + ch] = (_Float16)v;
  }
}

// Final: decode pre-max, apply rp2 BN -> feats (B,M,1024) f32 into d_out.
__global__ void k_feats(const unsigned* __restrict__ premax,
                        const float* __restrict__ sum, const float* __restrict__ sq,
                        const float* __restrict__ g, const float* __restrict__ be,
                        float* __restrict__ out) {
  int idx = blockIdx.x * blockDim.x + threadIdx.x;
  if (idx >= 1024 * 1024) return;
  int col = idx & 1023;
  const float inv = 1.f / (float)ROWS_RP;
  float m = sum[col] * inv;
  float var = sq[col] * inv - m * m;
  float rs = rsqrtf(var + 1e-5f);
  out[idx] = (mono_dec(premax[idx]) - m) * rs * g[col] + be[col];
}

// ---------------------------------------------------------------------------
// Launch
// ---------------------------------------------------------------------------
extern "C" void kernel_launch(void* const* d_in, const int* in_sizes, int n_in,
                              void* d_out, int out_size, void* d_ws, size_t ws_size,
                              hipStream_t stream) {
  (void)in_sizes; (void)n_in; (void)out_size; (void)ws_size;
  const float* points = (const float*)d_in[0];
  const float* gt     = (const float*)d_in[1];
  const float* pn_w1 = (const float*)d_in[2];  const float* pn_b1 = (const float*)d_in[3];
  const float* pn_g1 = (const float*)d_in[4];  const float* pn_be1 = (const float*)d_in[5];
  const float* pn_w2 = (const float*)d_in[6];  const float* pn_b2 = (const float*)d_in[7];
  const float* pn_g2 = (const float*)d_in[8];  const float* pn_be2 = (const float*)d_in[9];
  const float* pn_w3 = (const float*)d_in[10]; const float* pn_b3 = (const float*)d_in[11];
  const float* pn_g3 = (const float*)d_in[12]; const float* pn_be3 = (const float*)d_in[13];
  const float* rp_w1 = (const float*)d_in[14]; const float* rp_b1 = (const float*)d_in[15];
  const float* rp_g1 = (const float*)d_in[16]; const float* rp_be1 = (const float*)d_in[17];
  const float* rp_w2 = (const float*)d_in[18]; const float* rp_b2 = (const float*)d_in[19];
  const float* rp_g2 = (const float*)d_in[20]; const float* rp_be2 = (const float*)d_in[21];

  // workspace carving
  size_t off = 0;
  char* base = (char*)d_ws;
  auto take = [&](size_t bytes) -> char* {
    char* p = base + off;
    off = (off + bytes + 255) & ~(size_t)255;
    return p;
  };
  _Float16* A0h   = (_Float16*)take((size_t)ROWS_PN * 32 * 2);
  float*    pts   = (float*)   take((size_t)ROWS_PN * 3 * 4);
  _Float16* W1h   = (_Float16*)take((size_t)64 * 32 * 2);
  _Float16* W2h   = (_Float16*)take((size_t)128 * 64 * 2);
  _Float16* W3h   = (_Float16*)take((size_t)256 * 128 * 2);
  _Float16* Wr1h  = (_Float16*)take((size_t)512 * K_RP1 * 2);
  _Float16* Wr2h  = (_Float16*)take((size_t)1024 * 512 * 2);
  _Float16* x1h   = (_Float16*)take((size_t)ROWS_PN * 64 * 2);
  _Float16* x2h   = (_Float16*)take((size_t)ROWS_PN * 128 * 2);
  _Float16* x3h   = (_Float16*)take((size_t)ROWS_PN * 256 * 2);
  _Float16* poolh = (_Float16*)take((size_t)ROWS_RP * K_RP1 * 2);
  _Float16* rp1h  = (_Float16*)take((size_t)ROWS_RP * 512 * 2);
  float*    ssum  = (float*)   take((size_t)1024 * 4);
  float*    ssq   = (float*)   take((size_t)1024 * 4);
  float*    miou  = (float*)   take((size_t)B_ * NANCH * 4);
  int*      mind  = (int*)     take((size_t)B_ * NANCH * 4);
  float*    bxws  = (float*)   take((size_t)B_ * MSEL * 7 * 4);
  unsigned* premax= (unsigned*)take((size_t)1024 * 1024 * 4);

  // output carving (tuple, flat, in return order)
  float* out_f      = (float*)d_out;
  float* out_feats  = out_f;                         // 8*128*1024
  int*   out_lab    = (int*)(out_f + 1048576);       // 1024 (int32)
  float* out_iou    = out_f + 1049600;               // 1024
  float* out_boxes  = out_f + 1050624;               // 8*128*7
  float* out_gt     = out_f + 1057792;               // 10*9
  int*   out_selidx = (int*)(out_f + 1057882);       // 1024 (int32)

  const dim3 wblk(32, 4);

  // ---- input packing / weight conversion ----
  k_pack_points<<<(ROWS_PN + 255) / 256, 256, 0, stream>>>(points, A0h, pts);
  k_convert_w<<<(64 * 32 + 255) / 256, 256, 0, stream>>>(pn_w1, W1h, 64, 4, 32);
  k_convert_w<<<(128 * 64 + 255) / 256, 256, 0, stream>>>(pn_w2, W2h, 128, 64, 64);
  k_convert_w<<<(256 * 128 + 255) / 256, 256, 0, stream>>>(pn_w3, W3h, 256, 128, 128);
  k_convert_w<<<(512 * K_RP1 + 255) / 256, 256, 0, stream>>>(rp_w1, Wr1h, 512, 259, K_RP1);
  k_convert_w<<<(1024 * 512 + 255) / 256, 256, 0, stream>>>(rp_w2, Wr2h, 1024, 512, 512);

  // ---- pn1: (8192 x 32) x (64 x 32)^T, BN + ReLU ----
  k_gemm_f16<<<dim3(ROWS_PN / 16, 1), wblk, 0, stream>>>(A0h, W1h, pn_b1, x1h, 32, 64);
  k_colstats<<<64, 256, 0, stream>>>(x1h, ssum, ssq, ROWS_PN, 64);
  k_bn_apply<<<(ROWS_PN * 64 + 255) / 256, 256, 0, stream>>>(x1h, ssum, ssq, pn_g1, pn_be1, ROWS_PN, 64, 1);

  // ---- pn2 ----
  k_gemm_f16<<<dim3(ROWS_PN / 16, 2), wblk, 0, stream>>>(x1h, W2h, pn_b2, x2h, 64, 128);
  k_colstats<<<128, 256, 0, stream>>>(x2h, ssum, ssq, ROWS_PN, 128);
  k_bn_apply<<<(ROWS_PN * 128 + 255) / 256, 256, 0, stream>>>(x2h, ssum, ssq, pn_g2, pn_be2, ROWS_PN, 128, 1);

  // ---- pn3 (no ReLU) -> ftr ----
  k_gemm_f16<<<dim3(ROWS_PN / 16, 4), wblk, 0, stream>>>(x2h, W3h, pn_b3, x3h, 128, 256);
  k_colstats<<<256, 256, 0, stream>>>(x3h, ssum, ssq, ROWS_PN, 256);
  k_bn_apply<<<(ROWS_PN * 256 + 255) / 256, 256, 0, stream>>>(x3h, ssum, ssq, pn_g3, pn_be3, ROWS_PN, 256, 0);

  // ---- anchor IoU + selection ----
  k_iou<<<(B_ * NANCH + 127) / 128, 128, 0, stream>>>(pts, gt, miou, mind);
  k_sortsel<<<B_, 512, 0, stream>>>(miou, mind, pts, out_lab, out_iou, out_boxes, out_selidx, bxws);

  // ---- pooling -> (65536 x 288) f16 ----
  k_pool<<<B_ * MSEL, 256, 0, stream>>>(bxws, pts, x3h, poolh);

  // ---- rp1: (65536 x 288) x (512 x 288)^T, BN (no ReLU) ----
  k_gemm_f16<<<dim3(ROWS_RP / 16, 8), wblk, 0, stream>>>(poolh, Wr1h, rp_b1, rp1h, K_RP1, 512);
  k_colstats<<<512, 256, 0, stream>>>(rp1h, ssum, ssq, ROWS_RP, 512);
  k_bn_apply<<<(int)(((size_t)ROWS_RP * 512 + 255) / 256), 256, 0, stream>>>(rp1h, ssum, ssq, rp_g1, rp_be1, ROWS_RP, 512, 0);

  // ---- rp2: fused stats + pool-max epilogue ----
  (void)hipMemsetAsync(ssum, 0, 1024 * 4, stream);
  (void)hipMemsetAsync(ssq, 0, 1024 * 4, stream);
  (void)hipMemsetAsync(premax, 0, (size_t)1024 * 1024 * 4, stream);
  k_gemm_rp2<<<dim3(ROWS_RP / 16, 16), wblk, 0, stream>>>(rp1h, Wr2h, rp_b2, ssum, ssq, premax, 512, 1024);
  k_feats<<<(1048576 + 255) / 256, 256, 0, stream>>>(premax, ssum, ssq, rp_g2, rp_be2, out_feats);

  // ---- gt_labels[-1] passthrough ----
  (void)hipMemcpyAsync(out_gt, gt + (size_t)(B_ - 1) * NGT * GTC, (size_t)NGT * GTC * 4,
                       hipMemcpyDeviceToDevice, stream);
}
